// SSGCN_22591527977030
// MI455X (gfx1250) — matrix operand order repaired and checked by reading, not compiled, not run on hardware
//
#include <hip/hip_runtime.h>
#include <hip/hip_bf16.h>
#include <stdint.h>

// ---------------------------------------------------------------------------
// SSGCN on MI455X (gfx1250, wave32, WMMA)
//
//  1. zero node accumulators (2 x N floats)
//  2. fused edge scatter: ONE pass over 20.48M edges feeds BOTH encoders.
//     Edge stream (164MB, read once) is loaded non-temporally so the L2
//     keeps the 8MB accumulators + 8MB gather sources resident for the
//     41M float atomics (the true bottleneck of this net).
//  3. relu(agg + gcn_b) -> f16, zero-padded K 978->1024  (WMMA A operand)
//  4. pack fc1_w / fc2_w into WMMA B-fragment order (f16, zero-padded)
//  5. GEMM1 via v_wmma_f32_16x16x32_f16, 2Mx4N register blocking per wave
//     (8 wmma per 12 b128 loads), epilogue bias+ReLU -> f16 row-major
//  6. GEMM2, 1Mx7N blocking (A fragment reused 7x), epilogue bias -> f32
//  7. per-graph R^2 (wave shuffle reductions) + 5->4->2 MLP -> out (1024x2)
// ---------------------------------------------------------------------------

#define B_GRAPHS   1024
#define NG         978
#define NNODES     (B_GRAPHS * NG)        // 1,001,472
#define NEDGES     20480000
#define K1PAD      1024                    // 978 padded to 32-multiple
#define N1         2048
#define K2         2048
#define N2REAL     100
#define N2PAD      112                     // 100 padded to 16-multiple
#define K1TILES    (K1PAD / 32)            // 32
#define K2TILES    (K2 / 32)               // 64
#define N1TILES    (N1 / 16)               // 128
#define N2TILES    (N2PAD / 16)            // 7

typedef _Float16 v16h __attribute__((ext_vector_type(16)));
typedef float    v8f  __attribute__((ext_vector_type(8)));

union AFrag { v16h v; uint32_t u[8]; };
union BFrag { v16h v; uint4    q[2]; };

// ---------------- phase 1: zero accumulators --------------------------------
__global__ __launch_bounds__(256) void zero_f32(float* p, int n) {
    int i = blockIdx.x * 256 + threadIdx.x;
    int stride = gridDim.x * 256;
    for (; i < n; i += stride) __builtin_nontemporal_store(0.0f, &p[i]);
}

// ---------------- phase 2: fused dual-input GCN scatter ---------------------
__global__ __launch_bounds__(256) void gcn_scatter(const float* __restrict__ x1,
                                                   const float* __restrict__ x2,
                                                   const int*   __restrict__ src,
                                                   const int*   __restrict__ dst,
                                                   const float* __restrict__ gw,
                                                   float* __restrict__ agg1,
                                                   float* __restrict__ agg2) {
    const float w = *gw;
    int i = blockIdx.x * 256 + threadIdx.x;
    const int stride = gridDim.x * 256;
    for (; i < NEDGES; i += stride) {
        int pf = i + stride;
        if (pf < NEDGES) {                      // global_prefetch_b8 on edge stream
            __builtin_prefetch(&src[pf], 0, 1);
            __builtin_prefetch(&dst[pf], 0, 1);
        }
        // edge list: 164MB streamed once -> non-temporal, keep L2 for aggs/x
        const int s = __builtin_nontemporal_load(&src[i]);
        const int d = __builtin_nontemporal_load(&dst[i]);
        atomicAdd(&agg1[d], x1[s] * w);
        atomicAdd(&agg2[d], x2[s] * w);
    }
}

// ---------------- phase 3: relu + convert + pad A1 --------------------------
__global__ __launch_bounds__(256) void relu_cvt_a(const float* __restrict__ agg,
                                                  const float* __restrict__ gbias,
                                                  _Float16* __restrict__ A) {
    int idx = blockIdx.x * 256 + threadIdx.x;       // 1024*1024 threads
    if (idx >= B_GRAPHS * K1PAD) return;
    const int r = idx >> 10;          // / K1PAD
    const int c = idx & (K1PAD - 1);
    float v = 0.0f;
    if (c < NG) {
        v = agg[r * NG + c] + *gbias;
        v = v > 0.0f ? v : 0.0f;
    }
    A[idx] = (_Float16)v;
}

// ---------------- phase 4: pack weights into B-fragment order ---------------
// Wp[((tileN*Ktiles + tileK)*32 + lane)*16 + e], lane: col = tN*16 + lane%16,
// khi = lane/16; element e: k = tK*32 + 8*khi + (e&7) + 16*(e>>3)  (ISA 16-bit B 32x16 layout)
__global__ __launch_bounds__(256) void pack_b(const float* __restrict__ W,
                                              _Float16* __restrict__ Wp,
                                              int Kreal, int Nreal,
                                              int Ktiles, int Ntiles) {
    int idx = blockIdx.x * 256 + threadIdx.x;
    const int total = Ntiles * Ktiles * 512;
    if (idx >= total) return;
    const int e    = idx & 15;
    const int lane = (idx >> 4) & 31;
    const int t    = idx >> 9;
    const int tk   = t % Ktiles;
    const int tn   = t / Ktiles;
    const int col  = tn * 16 + (lane & 15);
    const int k    = tk * 32 + ((lane >> 4) << 3) + (e & 7) + ((e >> 3) << 4);
    float v = (k < Kreal && col < Nreal) ? W[k * Nreal + col] : 0.0f;
    Wp[idx] = (_Float16)v;
}

// ---------------- WMMA fragment loads ---------------------------------------
// A (16-bit 16x32): lane holds row M=lane%16; VGPR v pairs K = {2(v&3) +
// 16*(v>>2) + 8*(lane>=16), +1} -> two contiguous 4-dword groups -> 2x b128.
__device__ __forceinline__ v16h load_a_frag(const _Float16* __restrict__ A,
                                            int row, int k0, int lda, int lane) {
    const int laneHi = lane >> 4;
    const uint32_t* A32 = (const uint32_t*)A;
    const int base = row * lda + k0 + 8 * laneHi;
    AFrag f;
#pragma unroll
    for (int v = 0; v < 8; ++v) {
        const int koff = 2 * (v & 3) + ((v >> 2) << 4);
        f.u[v] = A32[(base + koff) >> 1];
    }
    return f.v;
}

// B pre-swizzled: 32 contiguous bytes per lane per k-tile -> two b128 loads.
__device__ __forceinline__ v16h load_b_packed(const _Float16* __restrict__ Wp,
                                              int tileIdx, int lane) {
    const uint4* p = (const uint4*)(Wp + (size_t)tileIdx * 512 + lane * 16);
    BFrag f;
    f.q[0] = p[0];
    f.q[1] = p[1];
    return f.v;
}

__device__ __forceinline__ v8f wmma_f16(v16h a, v16h b, v8f c) {
    return __builtin_amdgcn_wmma_f32_16x16x32_f16(false, a, false, b,
                                                  (short)0, c, false, false);
}

// ---------------- phase 5: GEMM1 + bias + ReLU -> f16 -----------------------
// Wave computes a 32x64 output block (2 M-tiles x 4 N-tiles): A fragments
// reused 4x, B fragments reused 2x -> 8 wmma per 12 b128 loads per k-step.
// grid (32, 4), block (32,8): wave w -> mGroup = bx, nGroup = by*8 + w.
__global__ __launch_bounds__(256) void gemm1_wmma(const _Float16* __restrict__ A,
                                                  const _Float16* __restrict__ Wp,
                                                  const float* __restrict__ bias,
                                                  _Float16* __restrict__ H) {
    const int lane = threadIdx.x;
    const int mg   = blockIdx.x;                    // 0..31 -> tileM = mg*2+{0,1}
    const int ng   = blockIdx.y * 8 + threadIdx.y;  // 0..31 -> tileN = ng*4+{0..3}
    const int row0 = mg * 32 + (lane & 15);

    v8f acc[8] = {};                                // [mi*4 + ni]
#pragma unroll 2
    for (int kt = 0; kt < K1TILES; ++kt) {
        v16h a0 = load_a_frag(A, row0,      kt * 32, K1PAD, lane);
        v16h a1 = load_a_frag(A, row0 + 16, kt * 32, K1PAD, lane);
#pragma unroll
        for (int ni = 0; ni < 4; ++ni) {
            v16h b = load_b_packed(Wp, (ng * 4 + ni) * K1TILES + kt, lane);
            acc[ni]     = wmma_f16(a0, b, acc[ni]);
            acc[4 + ni] = wmma_f16(a1, b, acc[4 + ni]);
        }
    }
    // C layout: VGPR v -> M = v + 8*(lane>=16), N = lane%16
    const int laneHi = lane >> 4;
#pragma unroll
    for (int ni = 0; ni < 4; ++ni) {
        const int n  = (ng * 4 + ni) * 16 + (lane & 15);
        const float bn = bias[n];
#pragma unroll
        for (int mi = 0; mi < 2; ++mi) {
#pragma unroll
            for (int v = 0; v < 8; ++v) {
                const int m = (mg * 2 + mi) * 16 + v + 8 * laneHi;
                float val = acc[mi * 4 + ni][v] + bn;
                val = val > 0.0f ? val : 0.0f;
                H[(size_t)m * N1 + n] = (_Float16)val;
            }
        }
    }
}

// ---------------- phase 6: GEMM2 + bias -> f32 ------------------------------
// Wave computes 16 x 112 (1 M-tile x all 7 N-tiles): A fragment reused 7x.
// grid (8), block (32,8): wave w -> tileM = bx*8 + w.
__global__ __launch_bounds__(256) void gemm2_wmma(const _Float16* __restrict__ H,
                                                  const _Float16* __restrict__ Wp,
                                                  const float* __restrict__ bias,
                                                  float* __restrict__ O) {
    const int lane  = threadIdx.x;
    const int tileM = blockIdx.x * 8 + threadIdx.y;  // 0..63
    const int row   = tileM * 16 + (lane & 15);

    v8f acc[N2TILES] = {};
#pragma unroll 2
    for (int kt = 0; kt < K2TILES; ++kt) {
        v16h a = load_a_frag(H, row, kt * 32, K2, lane);
#pragma unroll
        for (int n = 0; n < N2TILES; ++n) {
            v16h b = load_b_packed(Wp, n * K2TILES + kt, lane);
            acc[n] = wmma_f16(a, b, acc[n]);
        }
    }
    const int laneHi = lane >> 4;
#pragma unroll
    for (int n = 0; n < N2TILES; ++n) {
        const int col = n * 16 + (lane & 15);
        if (col < N2REAL) {
            const float bn = bias[col];
#pragma unroll
            for (int v = 0; v < 8; ++v) {
                const int m = tileM * 16 + v + 8 * laneHi;
                O[(size_t)m * N2REAL + col] = acc[n][v] + bn;
            }
        }
    }
}

// ---------------- phase 7: per-graph R^2 + MLP ------------------------------
__global__ __launch_bounds__(256) void r2_mlp(const float* __restrict__ O1,
                                              const float* __restrict__ O2,
                                              const float* __restrict__ others,
                                              const float* __restrict__ m1w,
                                              const float* __restrict__ m1b,
                                              const float* __restrict__ m2w,
                                              const float* __restrict__ m2b,
                                              float* __restrict__ out) {
    const int lane = threadIdx.x;       // 0..31
    const int g    = blockIdx.x * 8 + threadIdx.y;   // graph id, grid 128 x 8 waves
    const float* a = O1 + (size_t)g * N2REAL;
    const float* c = O2 + (size_t)g * N2REAL;

    float sa = 0.0f, sc = 0.0f;
    for (int i = lane; i < N2REAL; i += 32) { sa += a[i]; sc += c[i]; }
#pragma unroll
    for (int off = 16; off; off >>= 1) {
        sa += __shfl_xor(sa, off, 32);
        sc += __shfl_xor(sc, off, 32);
    }
    const float ma = sa * (1.0f / N2REAL);
    const float mc = sc * (1.0f / N2REAL);

    float sab = 0.0f, saa = 0.0f, scc = 0.0f;
    for (int i = lane; i < N2REAL; i += 32) {
        const float da = a[i] - ma;
        const float dc = c[i] - mc;
        sab += da * dc; saa += da * da; scc += dc * dc;
    }
#pragma unroll
    for (int off = 16; off; off >>= 1) {
        sab += __shfl_xor(sab, off, 32);
        saa += __shfl_xor(saa, off, 32);
        scc += __shfl_xor(scc, off, 32);
    }
    if (lane == 0) {
        const float den = __builtin_sqrtf(saa) * __builtin_sqrtf(scc);
        const float r   = sab / den;
        float z[5];
        z[0] = r * r;
        z[1] = others[g * 4 + 0];
        z[2] = others[g * 4 + 1];
        z[3] = others[g * 4 + 2];
        z[4] = others[g * 4 + 3];
        float h[4];
#pragma unroll
        for (int j = 0; j < 4; ++j) {
            float s = m1b[j];
#pragma unroll
            for (int i = 0; i < 5; ++i) s += z[i] * m1w[i * 4 + j];
            h[j] = s > 0.0f ? s : 0.0f;
        }
#pragma unroll
        for (int j = 0; j < 2; ++j) {
            float s = m2b[j];
#pragma unroll
            for (int i = 0; i < 4; ++i) s += h[i] * m2w[i * 2 + j];
            out[g * 2 + j] = s;
        }
    }
}

// ---------------------------------------------------------------------------
extern "C" void kernel_launch(void* const* d_in, const int* in_sizes, int n_in,
                              void* d_out, int out_size, void* d_ws, size_t ws_size,
                              hipStream_t stream) {
    const float* input1 = (const float*)d_in[0];
    const float* input2 = (const float*)d_in[1];
    const int*   edges  = (const int*)  d_in[2];
    const float* others = (const float*)d_in[3];
    const float* gcn_w  = (const float*)d_in[4];
    const float* gcn_b  = (const float*)d_in[5];
    const float* fc1_w  = (const float*)d_in[6];
    const float* fc1_b  = (const float*)d_in[7];
    const float* fc2_w  = (const float*)d_in[8];
    const float* fc2_b  = (const float*)d_in[9];
    const float* mlp1_w = (const float*)d_in[10];
    const float* mlp1_b = (const float*)d_in[11];
    const float* mlp2_w = (const float*)d_in[12];
    const float* mlp2_b = (const float*)d_in[13];
    float* out = (float*)d_out;

    const int* src = edges;
    const int* dst = edges + NEDGES;

    // workspace carve-up (256B aligned chunks)
    char* p = (char*)d_ws;
    auto carve = [&](size_t bytes) {
        void* r = (void*)p;
        p += (bytes + 255) & ~(size_t)255;
        return r;
    };
    float*    agg1 = (float*)   carve((size_t)NNODES * 4);
    float*    agg2 = (float*)   carve((size_t)NNODES * 4);
    _Float16* A1a  = (_Float16*)carve((size_t)B_GRAPHS * K1PAD * 2);
    _Float16* A1b  = (_Float16*)carve((size_t)B_GRAPHS * K1PAD * 2);
    _Float16* W1p  = (_Float16*)carve((size_t)N1TILES * K1TILES * 512 * 2);
    _Float16* H2a  = (_Float16*)carve((size_t)B_GRAPHS * N1 * 2);
    _Float16* H2b  = (_Float16*)carve((size_t)B_GRAPHS * N1 * 2);
    _Float16* W2p  = (_Float16*)carve((size_t)N2TILES * K2TILES * 512 * 2);
    float*    O1   = (float*)   carve((size_t)B_GRAPHS * N2REAL * 4);
    float*    O2   = (float*)   carve((size_t)B_GRAPHS * N2REAL * 4);
    (void)ws_size; (void)n_in; (void)in_sizes; (void)out_size;

    // 1. zero accumulators
    zero_f32<<<2048, 256, 0, stream>>>(agg1, NNODES);
    zero_f32<<<2048, 256, 0, stream>>>(agg2, NNODES);

    // 2. fused dual-input edge scatter
    gcn_scatter<<<4096, 256, 0, stream>>>(input1, input2, src, dst, gcn_w,
                                          agg1, agg2);

    // 3. relu + f32->f16 + K-pad for both encoders
    relu_cvt_a<<<(B_GRAPHS * K1PAD) / 256, 256, 0, stream>>>(agg1, gcn_b, A1a);
    relu_cvt_a<<<(B_GRAPHS * K1PAD) / 256, 256, 0, stream>>>(agg2, gcn_b, A1b);

    // 4. pack weights into WMMA B-fragment order
    {
        const int tot1 = N1TILES * K1TILES * 512;
        pack_b<<<(tot1 + 255) / 256, 256, 0, stream>>>(fc1_w, W1p, NG, N1,
                                                       K1TILES, N1TILES);
        const int tot2 = N2TILES * K2TILES * 512;
        pack_b<<<(tot2 + 255) / 256, 256, 0, stream>>>(fc2_w, W2p, K2, N2REAL,
                                                       K2TILES, N2TILES);
    }

    // 5. GEMM1 (WMMA, 2Mx4N blocking) + bias + ReLU -> f16
    {
        dim3 blk(32, 8), grd(32, 4);
        gemm1_wmma<<<grd, blk, 0, stream>>>(A1a, W1p, fc1_b, H2a);
        gemm1_wmma<<<grd, blk, 0, stream>>>(A1b, W1p, fc1_b, H2b);
    }

    // 6. GEMM2 (WMMA, 1Mx7N blocking) + bias -> f32
    {
        dim3 blk(32, 8), grd(8);
        gemm2_wmma<<<grd, blk, 0, stream>>>(H2a, W2p, fc2_b, O1);
        gemm2_wmma<<<grd, blk, 0, stream>>>(H2b, W2p, fc2_b, O2);
    }

    // 7. R^2 + MLP head
    r2_mlp<<<B_GRAPHS / 8, dim3(32, 8), 0, stream>>>(O1, O2, others,
                                                     mlp1_w, mlp1_b,
                                                     mlp2_w, mlp2_b, out);
}
